// SingleHeadAttention_29437705846902
// MI455X (gfx1250) — matrix operand order repaired and checked by reading, not compiled, not run on hardware
//
#include <hip/hip_runtime.h>
#include <hip/hip_bf16.h>

// ---------------------------------------------------------------------------
// Single-head causal attention for MI455X (gfx1250, wave32, WMMA + TDM).
//   proj:  X(16384x256) @ W^T(256x256) -> f16 (Qp/Kp row-major, Vt transposed)
//   flash: fused QK^T -> online softmax -> PV, f16 WMMA, f32 accum,
//          causal chunk skip, K/V chunks staged per-block into double-buffered
//          LDS by the Tensor Data Mover (with hardware row padding so WMMA
//          B-operand ds_load_b128s are bank-conflict free).
// ---------------------------------------------------------------------------

#define D_MODEL 256
#define BATCH   4
#define SEQ     4096

// padded LDS row strides (in halves): +8 halves (16B) kills bank conflicts
#define QK_STRIDE 264   // 256 + 8  (512B + 16B)
#define V_STRIDE  40    // 32 + 8   (64B + 16B)
#define P_STRIDE  40    // 32 + 8

typedef _Float16 v16h  __attribute__((ext_vector_type(16)));
typedef _Float16 half8 __attribute__((ext_vector_type(8)));
typedef float    v8f   __attribute__((ext_vector_type(8)));
typedef float    v4f   __attribute__((ext_vector_type(4)));
typedef float    f8v   __attribute__((ext_vector_type(8)));
typedef unsigned int u32x4 __attribute__((ext_vector_type(4)));
typedef int          i32x8 __attribute__((ext_vector_type(8)));
typedef int          i32x4 __attribute__((ext_vector_type(4)));

#if defined(__has_builtin)
#if __has_builtin(__builtin_amdgcn_tensor_load_to_lds) && \
    __has_builtin(__builtin_amdgcn_s_wait_tensorcnt)
#define USE_TDM 1
#endif
#endif
#ifndef USE_TDM
#define USE_TDM 0
#endif

__device__ __forceinline__ v16h cat16(half8 lo, half8 hi) {
    return __builtin_shufflevector(lo, hi, 0,1,2,3,4,5,6,7,8,9,10,11,12,13,14,15);
}
__device__ __forceinline__ half8 cvt8(const float* p) {
    v4f x0 = *(const v4f*)p;
    v4f x1 = *(const v4f*)(p + 4);
    f8v f  = __builtin_shufflevector(x0, x1, 0,1,2,3,4,5,6,7);
    return __builtin_convertvector(f, half8);
}

typedef __attribute__((address_space(3))) _Float16 lds_f16;
__device__ __forceinline__ unsigned ldsOffset(_Float16* p) {
    return (unsigned)(unsigned long long)(lds_f16*)p;
}

#if USE_TDM
// Issue one TDM 2D tile load (global -> LDS) with optional LDS row padding.
// Descriptor layout per CDNA5 ISA sec. 8.3/8.4 (D# groups 0/1; trailing
// groups zero => <=2D tensor).  data_size=2B.  pad fields give padded rows.
// This toolchain exposes the 6-arg form:
//   (u32x4 g0, i32x8 g1, i32x4, i32x4, i32x8, i32 cpol)
__device__ __forceinline__ void tdm_issue(unsigned ldsOff, const void* gsrc,
                                          unsigned long long tensorX, unsigned tensorY,
                                          unsigned tileX, unsigned tileY,
                                          unsigned long long stride0,
                                          unsigned padI, unsigned padA)
{
    const unsigned long long ga = (unsigned long long)gsrc;
    u32x4 g0;
    g0[0] = 1u;                                        // count=1 (valid), user mode
    g0[1] = ldsOff;                                    // lds_addr (bytes)
    g0[2] = (unsigned)ga;                              // global_addr[31:0]
    g0[3] = (unsigned)((ga >> 32) & 0x1FFFFFFu)        // global_addr[56:32]
          | (2u << 30);                                // type=2 ("image")
    i32x8 g1;
    g1[0] = (int)((1u << 16)                           // data_size=1 -> 2 bytes
                | (1u << 20)                           // pad_enable
                | (padI << 22) | (padA << 25));        // pad_interval / pad_amount
    const unsigned tX = (unsigned)tensorX;
    g1[1] = (int)((tX & 0xFFFFu) << 16);               // tensor_dim0[15:0]
    g1[2] = (int)((tX >> 16) | ((tensorY & 0xFFFFu) << 16));
    g1[3] = (int)(((tensorY >> 16) & 0xFFFFu) | ((tileX & 0xFFFFu) << 16));
    g1[4] = (int)(tileY & 0xFFFFu);                    // tile_dim1 (tile_dim2=0)
    g1[5] = (int)(stride0 & 0xFFFFFFFFull);            // tensor_dim0_stride[31:0]
    g1[6] = (int)((stride0 >> 32) & 0xFFFFull);
    g1[7] = 0;
    i32x4 z4 = {0, 0, 0, 0};
    i32x8 z8 = {0, 0, 0, 0, 0, 0, 0, 0};
    __builtin_amdgcn_tensor_load_to_lds(g0, g1, z4, z4, z8, 0);
}
#endif

// Stage one 32-key chunk of K (32 x 256 f16, padded rows) and V (256 x 32 f16,
// padded rows) into LDS.  TDM path: wave0 issues two descriptors (async,
// TENSORcnt).  Fallback: synchronous cooperative copy by all 128 threads.
__device__ __forceinline__ void stage_chunk(int b, int s0,
                                            _Float16* kDst, _Float16* vDst,
                                            const _Float16* __restrict__ Kp,
                                            const _Float16* __restrict__ Vt,
                                            int tid)
{
#if USE_TDM
    if (tid < 32) {
        // K chunk: 16KB contiguous, pad 16B after every 512B row.
        tdm_issue(ldsOffset(kDst), Kp + (size_t)(b * SEQ + s0) * D_MODEL,
                  /*tensorX=*/32 * D_MODEL, /*tensorY=*/1,
                  /*tileX=*/32 * D_MODEL, /*tileY=*/1,
                  /*stride0=*/32 * D_MODEL, /*padI=*/6, /*padA=*/3);
        // V chunk: 256 rows of 64B strided by SEQ, pad 16B after every 64B row.
        tdm_issue(ldsOffset(vDst), Vt + (size_t)b * D_MODEL * SEQ + s0,
                  /*tensorX=*/SEQ, /*tensorY=*/D_MODEL,
                  /*tileX=*/32, /*tileY=*/D_MODEL,
                  /*stride0=*/SEQ, /*padI=*/3, /*padA=*/3);
    }
#else
    for (int i = tid; i < 1024; i += 128) {           // 32 rows x 32 b128
        const int row = i >> 5, part = i & 31;
        *(uint4*)&kDst[row * QK_STRIDE + part * 8] =
            *(const uint4*)(Kp + (size_t)(b * SEQ + s0 + row) * D_MODEL + part * 8);
    }
    for (int i = tid; i < 1024; i += 128) {           // 256 rows x 4 b128
        const int row = i >> 2, part = i & 3;
        *(uint4*)&vDst[row * V_STRIDE + part * 8] =
            *(const uint4*)(Vt + ((size_t)b * D_MODEL + row) * SEQ + s0 + part * 8);
    }
#endif
}

__device__ __forceinline__ void stage_wait_keep2(int tid) {
#if USE_TDM
    if (tid < 32) __builtin_amdgcn_s_wait_tensorcnt(2);  // prev chunk done
#else
    (void)tid;
#endif
}
__device__ __forceinline__ void stage_wait_all(int tid) {
#if USE_TDM
    if (tid < 32) __builtin_amdgcn_s_wait_tensorcnt(0);
#else
    (void)tid;
#endif
}

// ---------------------------------------------------------------------------
// Projection: out[s,e] = sum_d x[s,d] * W[e,d], f32 in -> f16 out.
// One 16x16 output tile per wave via v_wmma_f32_16x16x32_f16.
// ---------------------------------------------------------------------------
__global__ __launch_bounds__(128, 2)
void proj_kernel(const float* __restrict__ x, const float* __restrict__ W,
                 _Float16* __restrict__ out, int transposeOut)
{
    const int wave = threadIdx.x >> 5;
    const int lane = threadIdx.x & 31;
    const int half = lane >> 4;
    const int l16  = lane & 15;

    const int m0 = (blockIdx.x * 4 + wave) * 16;   // row tile (over B*S)
    const int n0 = blockIdx.y * 16;                // col tile (over D_MODEL)

    v8f c = {};
#pragma unroll
    for (int kc = 0; kc < 8; ++kc) {
        const int k0 = kc * 32;
        const float* ap = x + (size_t)(m0 + l16) * D_MODEL + k0 + half * 8;
        v16h A = cat16(cvt8(ap), cvt8(ap + 16));
        const float* bp = W + (size_t)(n0 + l16) * D_MODEL + k0 + half * 8;
        v16h Bm = cat16(cvt8(bp), cvt8(bp + 16));
        c = __builtin_amdgcn_wmma_f32_16x16x32_f16(false, A, false, Bm,
                                                   (short)0, c, false, false);
    }
#pragma unroll
    for (int r = 0; r < 8; ++r) {
        const int row = m0 + half * 8 + r;
        const int col = n0 + l16;
        const _Float16 v = (_Float16)c[r];
        if (transposeOut) {
            const int bb = row >> 12;        // row / SEQ
            const int s  = row & (SEQ - 1);  // row % SEQ
            out[((size_t)bb * D_MODEL + col) * SEQ + s] = v;
        } else {
            out[(size_t)row * D_MODEL + col] = v;
        }
    }
}

// ---------------------------------------------------------------------------
// Fused causal flash attention.  Block = 4 waves / 64 query rows; each wave
// owns a 16-row Q tile with full D=256 f32 accumulation (16 WMMA C tiles).
// K/V chunks are TDM-staged once per block into double-buffered padded LDS.
// ---------------------------------------------------------------------------
__global__ __launch_bounds__(128, 1)
void flash_kernel(const _Float16* __restrict__ Qp, const _Float16* __restrict__ Kp,
                  const _Float16* __restrict__ Vt, float* __restrict__ out)
{
    const int wave = threadIdx.x >> 5;
    const int lane = threadIdx.x & 31;
    const int half = lane >> 4;
    const int l16  = lane & 15;

    const int qBlocks = SEQ / 64;
    const int b     = blockIdx.x / qBlocks;
    const int qBase = (blockIdx.x % qBlocks) * 64;
    const int q0    = qBase + wave * 16;

    __shared__ _Float16 qLds[4][16 * QK_STRIDE];   // ~8.25KB / wave
    __shared__ _Float16 pLds[4][16 * P_STRIDE];    // 1.25KB / wave
    __shared__ _Float16 kBuf[2][32 * QK_STRIDE];   // 16.5KB / buffer
    __shared__ _Float16 vBuf[2][D_MODEL * V_STRIDE]; // 20KB / buffer

    // Stage this wave's 16x256 Q tile into padded LDS rows.
    {
        const _Float16* src = Qp + (size_t)(b * SEQ + q0) * D_MODEL;
        for (int i = lane; i < 512; i += 32) {
            const int row = i >> 5, part = i & 31;
            *(uint4*)&qLds[wave][row * QK_STRIDE + part * 8] =
                *(const uint4*)(src + row * D_MODEL + part * 8);
        }
    }
    asm volatile("s_wait_dscnt 0" ::: "memory");

    v8f acc[16];
#pragma unroll
    for (int nt = 0; nt < 16; ++nt) acc[nt] = v8f{};
    float m_r[8], l_r[8];
#pragma unroll
    for (int r = 0; r < 8; ++r) { m_r[r] = -__builtin_inff(); l_r[r] = 0.f; }

    const float scale = 0.0625f;                 // 1/sqrt(256)
    const float L2E   = 1.44269504089f;
    const int myChunks = (q0 + 16 + 31) >> 5;    // this wave's causal bound
    const int nMax     = (qBase + 48 + 16 + 31) >> 5;  // block-uniform bound

    stage_chunk(b, 0, kBuf[0], vBuf[0], Kp, Vt, threadIdx.x);

    for (int c = 0; c < nMax; ++c) {
        const int s0 = c * 32;
        const int bi = c & 1;
        if (c + 1 < nMax) {
            stage_chunk(b, s0 + 32, kBuf[bi ^ 1], vBuf[bi ^ 1], Kp, Vt, threadIdx.x);
            stage_wait_keep2(threadIdx.x);       // chunk c landed, c+1 in flight
        } else {
            stage_wait_all(threadIdx.x);
        }
        __syncthreads();                         // publish chunk c to all waves

        if (c < myChunks) {
            // ---- logits: two 16x16 score tiles over keys [s0, s0+32) ----
            v8f sc[2]; sc[0] = v8f{}; sc[1] = v8f{};
#pragma unroll
            for (int kc = 0; kc < 8; ++kc) {
                const int k0 = kc * 32;
                const _Float16* ap = &qLds[wave][l16 * QK_STRIDE + k0 + half * 8];
                v16h A = cat16(*(const half8*)ap, *(const half8*)(ap + 16));
#pragma unroll
                for (int t = 0; t < 2; ++t) {
                    const _Float16* kp = &kBuf[bi][(t * 16 + l16) * QK_STRIDE
                                                   + k0 + half * 8];
                    v16h Bm = cat16(*(const half8*)kp, *(const half8*)(kp + 16));
                    sc[t] = __builtin_amdgcn_wmma_f32_16x16x32_f16(
                                false, A, false, Bm, (short)0, sc[t], false, false);
                }
            }

            // ---- online softmax (row stats live in the C-layout halves) ----
#pragma unroll
            for (int r = 0; r < 8; ++r) {
                const int rowg = q0 + half * 8 + r;
                float s0v = sc[0][r] * scale;
                float s1v = sc[1][r] * scale;
                if (s0 + l16      > rowg) s0v = -__builtin_inff();
                if (s0 + 16 + l16 > rowg) s1v = -__builtin_inff();

                float mx = fmaxf(s0v, s1v);
#pragma unroll
                for (int off = 8; off; off >>= 1)
                    mx = fmaxf(mx, __shfl_xor(mx, off, 32));
                const float mNew = fmaxf(m_r[r], mx);
                const float corr = exp2f((m_r[r] - mNew) * L2E);

                const float p0 = exp2f((s0v - mNew) * L2E);
                const float p1 = exp2f((s1v - mNew) * L2E);
                float sum = p0 + p1;
#pragma unroll
                for (int off = 8; off; off >>= 1)
                    sum += __shfl_xor(sum, off, 32);

                l_r[r] = l_r[r] * corr + sum;
                m_r[r] = mNew;
#pragma unroll
                for (int nt = 0; nt < 16; ++nt) acc[nt][r] *= corr;

                pLds[wave][(half * 8 + r) * P_STRIDE + l16]      = (_Float16)p0;
                pLds[wave][(half * 8 + r) * P_STRIDE + 16 + l16] = (_Float16)p1;
            }
            asm volatile("s_wait_dscnt 0" ::: "memory");

            // ---- P (16x32) as A operand ----
            const _Float16* pp = &pLds[wave][l16 * P_STRIDE + half * 8];
            v16h Pa = cat16(*(const half8*)pp, *(const half8*)(pp + 16));

            // ---- O += P @ V over 16 column tiles of D ----
#pragma unroll
            for (int nt = 0; nt < 16; ++nt) {
                const _Float16* vp = &vBuf[bi][(nt * 16 + l16) * V_STRIDE
                                               + half * 8];
                v16h Bv = cat16(*(const half8*)vp, *(const half8*)(vp + 16));
                acc[nt] = __builtin_amdgcn_wmma_f32_16x16x32_f16(
                              false, Pa, false, Bv, (short)0, acc[nt], false, false);
            }
        }
        __syncthreads();                         // buffer reuse fence
    }

    // ---- normalize and store f32 output ----
#pragma unroll
    for (int nt = 0; nt < 16; ++nt) {
#pragma unroll
        for (int r = 0; r < 8; ++r) {
            const int rowg = q0 + half * 8 + r;
            out[((size_t)b * SEQ + rowg) * D_MODEL + nt * 16 + l16] =
                acc[nt][r] / l_r[r];
        }
    }
}

// ---------------------------------------------------------------------------
extern "C" void kernel_launch(void* const* d_in, const int* in_sizes, int n_in,
                              void* d_out, int out_size, void* d_ws, size_t ws_size,
                              hipStream_t stream) {
    (void)in_sizes; (void)n_in; (void)out_size; (void)ws_size;
    const float* k  = (const float*)d_in[0];
    const float* q  = (const float*)d_in[1];
    const float* v  = (const float*)d_in[2];
    const float* Wk = (const float*)d_in[3];
    const float* Wq = (const float*)d_in[4];
    const float* Wv = (const float*)d_in[5];
    // d_in[6] (64MB causal mask) intentionally unread: causality is analytic.
    float* out = (float*)d_out;

    const size_t N = (size_t)BATCH * SEQ * D_MODEL;
    _Float16* Qp = (_Float16*)d_ws;       //  8 MB
    _Float16* Kp = Qp + N;                //  8 MB
    _Float16* Vt = Kp + N;                //  8 MB (per-batch transposed)

    dim3 pGrid((BATCH * SEQ / 16) / 4, D_MODEL / 16);
    proj_kernel<<<pGrid, 128, 0, stream>>>(q, Wq, Qp, 0);
    proj_kernel<<<pGrid, 128, 0, stream>>>(k, Wk, Kp, 0);
    proj_kernel<<<pGrid, 128, 0, stream>>>(v, Wv, Vt, 1);

    flash_kernel<<<BATCH * (SEQ / 64), 128, 0, stream>>>(Qp, Kp, Vt, out);
}
